// SRGNNRecommender_16947940950530
// MI455X (gfx1250) — matrix operand order repaired and checked by reading, not compiled, not run on hardware
//
#include <hip/hip_runtime.h>
#include <hip/hip_bf16.h>
#include <cstdint>

typedef __attribute__((ext_vector_type(16))) __bf16          v16bf;
typedef __attribute__((ext_vector_type(8)))  float           v8f;
typedef __attribute__((ext_vector_type(8)))  unsigned short  us8;   // 16B

#define BB 4096
#define NN 50
#define HH 128
#define M1 (BB * NN)   // 204800 rows

// ---------- bf16 helpers (RNE via bit trick; deterministic) ----------
__device__ __forceinline__ unsigned short f2bf(float f) {
  unsigned int u = __builtin_bit_cast(unsigned int, f);
  u = (u + 0x7FFFu + ((u >> 16) & 1u)) >> 16;
  return (unsigned short)u;
}
__device__ __forceinline__ float bf2f(unsigned short s) {
  unsigned int u = ((unsigned int)s) << 16;
  return __builtin_bit_cast(float, u);
}

union FragU { us8 h[2]; v16bf v; };

// ---------- WMMA fragment loaders ----------
// A: 16x32 bf16 tile from LDS (row-major [M][K]); each lane's 16 elements are
// one contiguous 16-element K run (k = 16*hi + s) -> two 16B LDS loads.
__device__ __forceinline__ v16bf load_a_lds(const unsigned short* As, int lda,
                                            int mBase, int kBase, int lane) {
  const int r = lane & 15, hi = (lane >> 4) & 1;
  const unsigned short* p = As + (mBase + r) * lda + kBase + hi * 16;
  FragU u;
  u.h[0] = *(const us8*)p;          // K = kBase + hi*16 + 0..7
  u.h[1] = *(const us8*)(p + 8);    // K = kBase + hi*16 + 8..15
  return u.v;
}
// B: pre-swizzled fragment storage. frag(jt,kt) = 512 contiguous ushorts,
// lane's 16 elements contiguous at lane*16. Works for global or LDS pointers.
__device__ __forceinline__ v16bf load_b_frag(const unsigned short* P, int fragIdx,
                                             int lane) {
  const unsigned short* p = P + ((size_t)fragIdx << 9) + lane * 16;
  FragU u;
  u.h[0] = *(const us8*)p;
  u.h[1] = *(const us8*)(p + 8);
  return u.v;
}
__device__ __forceinline__ v8f wmma_bf16(v16bf a, v16bf b, v8f c) {
  return __builtin_amdgcn_wmma_f32_16x16x32_bf16(false, a, false, b, (short)0, c,
                                                 false, false);
}

// =====================================================================
// K0: pack fp32 weight W[N][K] (w[j,k]) -> bf16 B-fragment order:
//   P[(jt*nkt + kt)*512 + (g*16 + nn)*16 + e] = bf( W[(jt*16+nn)*K + kt*32 + g*16 + e] )
// =====================================================================
__global__ __launch_bounds__(256) void k0_pack(const float* __restrict__ W,
                                               unsigned short* __restrict__ P,
                                               int Nrows, int K) {
  const int idx = blockIdx.x * 256 + threadIdx.x;
  if (idx >= Nrows * K) return;
  const int j = idx / K, k = idx - j * K;
  const int jt = j >> 4, nn = j & 15;
  const int kt = k >> 5, g = (k >> 4) & 1, e = k & 15;
  const int nkt = K >> 5;
  P[(((jt * nkt + kt) << 9) + (g * 16 + nn) * 16 + e)] = f2bf(W[idx]);
}

// =====================================================================
// K1: gather hidden = emb[items]; Hin/Hout = hidden * w^T + b.
// 64 rows/block, 8 waves; packed weights (nkt = 4).
// =====================================================================
__global__ __launch_bounds__(256) void k1_gather_dual_gemm(
    const int* __restrict__ items, const float* __restrict__ emb,
    const unsigned short* __restrict__ pWin, const float* __restrict__ b_in,
    const unsigned short* __restrict__ pWout, const float* __restrict__ b_out,
    unsigned short* __restrict__ hidB, unsigned short* __restrict__ HinB,
    unsigned short* __restrict__ HoutB) {
  __shared__ int itemS[64];
  __shared__ unsigned short Xs[64 * 128];     // 16 KB bf16 hidden tile (row-major)
  const int tid = threadIdx.x;
  const int m0 = blockIdx.x * 64;
  if (tid < 64) itemS[tid] = items[m0 + tid];
  __syncthreads();
  for (int idx = tid; idx < 64 * 128; idx += 256) {
    const int r = idx >> 7, c = idx & 127;
    const unsigned short s = f2bf(emb[(size_t)itemS[r] * HH + c]);
    Xs[idx] = s;
    hidB[(size_t)(m0 + r) * HH + c] = s;
  }
  __syncthreads();
  const int lane = tid & 31, wave = tid >> 5;
  const int n = lane & 15, g = (lane >> 4) & 1;
#pragma unroll
  for (int i = 0; i < 8; ++i) {
    const int t = wave * 8 + i;               // 0..63 ; mat uniform per wave
    const int mat = t >> 5, rest = t & 31, mt = rest >> 3, nt = rest & 7;
    const unsigned short* P = mat ? pWout : pWin;
    const float* bias = mat ? b_out : b_in;
    unsigned short* dst = mat ? HoutB : HinB;
    const float bv = bias[nt * 16 + n];
    v8f c;
#pragma unroll
    for (int e = 0; e < 8; ++e) c[e] = bv;
#pragma unroll
    for (int kt = 0; kt < 4; ++kt) {
      const v16bf a  = load_a_lds(Xs, 128, mt * 16, kt * 32, lane);
      const v16bf bb = load_b_frag(P, nt * 4 + kt, lane);
      c = wmma_bf16(a, bb, c);
    }
#pragma unroll
    for (int e = 0; e < 8; ++e) {
      const int row = m0 + mt * 16 + e + 8 * g;
      dst[(size_t)row * HH + nt * 16 + n] = f2bf(c[e]);
    }
  }
}

// =====================================================================
// K2: per-batch adjacency: inputs = [A_in @ Hin + b_iah, A_out @ Hout + b_oah].
// A tiles row-major (WMMA A); H tiles staged into LDS in B-fragment order.
// Frag index for H: kt*8 + nt (kt in 0..1, nt in 0..7).
// =====================================================================
__global__ __launch_bounds__(256) void k2_adjacency(
    const float* __restrict__ A,
    const unsigned short* __restrict__ HinB, const unsigned short* __restrict__ HoutB,
    const float* __restrict__ b_iah, const float* __restrict__ b_oah,
    unsigned short* __restrict__ inputsB) {
  __shared__ unsigned short AinS[64 * 64];    // 8 KB, row-major padded
  __shared__ unsigned short AoutS[64 * 64];   // 8 KB
  __shared__ unsigned short HinSw[16 * 512];  // 16 KB, B-frag order (2kt x 8nt)
  __shared__ unsigned short HoutSw[16 * 512]; // 16 KB
  const int tid = threadIdx.x;
  const int b = blockIdx.x;
  for (int idx = tid; idx < 64 * 64; idx += 256) { AinS[idx] = 0; AoutS[idx] = 0; }
  for (int idx = tid; idx < 16 * 512; idx += 256) { HinSw[idx] = 0; HoutSw[idx] = 0; }
  __syncthreads();
  for (int idx = tid; idx < NN * NN; idx += 256) {
    const int i = idx / NN, j = idx - i * NN;
    const float* ar = A + ((size_t)b * NN + i) * (2 * NN);
    AinS[i * 64 + j]  = f2bf(ar[j]);
    AoutS[i * 64 + j] = f2bf(ar[NN + j]);
  }
  for (int idx = tid; idx < NN * HH; idx += 256) {
    const int k = idx >> 7, c = idx & 127;    // k = H row (GEMM K), c = col (GEMM N)
    const size_t src = ((size_t)b * NN + k) * HH + c;
    const int dst = (((k >> 5) * 8 + (c >> 4)) << 9) +
                    ((((k >> 4) & 1) * 16 + (c & 15)) << 4) + (k & 15);
    HinSw[dst]  = HinB[src];
    HoutSw[dst] = HoutB[src];
  }
  __syncthreads();
  const int lane = tid & 31, wave = tid >> 5;
  const int n = lane & 15, g = (lane >> 4) & 1;
#pragma unroll
  for (int i = 0; i < 8; ++i) {
    const int t = wave * 8 + i;
    const int mat = t >> 5, rest = t & 31, mt = rest >> 3, nt = rest & 7;
    const unsigned short* As = mat ? AoutS : AinS;
    const unsigned short* Hs = mat ? HoutSw : HinSw;
    const float* bias = mat ? b_oah : b_iah;
    const float bv = bias[nt * 16 + n];
    v8f c;
#pragma unroll
    for (int e = 0; e < 8; ++e) c[e] = bv;
#pragma unroll
    for (int kt = 0; kt < 2; ++kt) {
      const v16bf a  = load_a_lds(As, 64, mt * 16, kt * 32, lane);
      const v16bf bb = load_b_frag(Hs, kt * 8 + nt, lane);
      c = wmma_bf16(a, bb, c);
    }
#pragma unroll
    for (int e = 0; e < 8; ++e) {
      const int rt = mt * 16 + e + 8 * g;
      if (rt < NN)
        inputsB[((size_t)b * NN + rt) * 256 + mat * 128 + nt * 16 + n] = f2bf(c[e]);
    }
  }
}

// =====================================================================
// K3: gi = inputs*w_ih^T + b_ih ; gh = hidden*w_hh^T + b_hh ; GRU gates.
// 16 rows/block. Packed weights: pWih nkt=8, pWhh nkt=4.
// =====================================================================
__global__ __launch_bounds__(256) void k3_gru(
    const unsigned short* __restrict__ inputsB, const unsigned short* __restrict__ hidB,
    const unsigned short* __restrict__ pWih, const unsigned short* __restrict__ pWhh,
    const float* __restrict__ b_ih, const float* __restrict__ b_hh,
    float* __restrict__ out) {
  __shared__ unsigned short hidS[16 * 128];   // 4 KB, persists
  __shared__ float gbuf[16 * 512];            // 32 KB: [0,4096)=ri [4096,6144)=gi_n [6144,8192)=gh_n
  unsigned short* inS = (unsigned short*)gbuf; // staging alias (first 8 KB)
  const int tid = threadIdx.x;
  const int m0 = blockIdx.x * 16;
  {  // vectorized staging (16B chunks)
    uint4* dI = (uint4*)inS;
    const uint4* sI = (const uint4*)(inputsB + (size_t)m0 * 256);
    for (int idx = tid; idx < 512; idx += 256) dI[idx] = sI[idx];
    uint4* dH = (uint4*)hidS;
    const uint4* sH = (const uint4*)(hidB + (size_t)m0 * 128);
    if (tid < 256) { if (tid < 256) dH[tid] = sH[tid]; }
  }
  __syncthreads();
  const int lane = tid & 31, wave = tid >> 5;
  const int n = lane & 15, g = (lane >> 4) & 1;
  v8f acc[4];
#pragma unroll
  for (int i = 0; i < 4; ++i) {
    const int t = wave + i * 8;               // wave-uniform class per i
    v8f c;
    if (t < 16) {                             // r,i gates: gi + gh fused, jt = t
      const int j0 = t * 16;
      const float bv = b_ih[j0 + n] + b_hh[j0 + n];
#pragma unroll
      for (int e = 0; e < 8; ++e) c[e] = bv;
#pragma unroll
      for (int kt = 0; kt < 8; ++kt)
        c = wmma_bf16(load_a_lds(inS, 256, 0, kt * 32, lane),
                      load_b_frag(pWih, t * 8 + kt, lane), c);
#pragma unroll
      for (int kt = 0; kt < 4; ++kt)
        c = wmma_bf16(load_a_lds(hidS, 128, 0, kt * 32, lane),
                      load_b_frag(pWhh, t * 4 + kt, lane), c);
    } else if (t < 24) {                      // gi_n, jt = 16 + (t-16)
      const int jt = 16 + (t - 16);
      const float bv = b_ih[jt * 16 + n];
#pragma unroll
      for (int e = 0; e < 8; ++e) c[e] = bv;
#pragma unroll
      for (int kt = 0; kt < 8; ++kt)
        c = wmma_bf16(load_a_lds(inS, 256, 0, kt * 32, lane),
                      load_b_frag(pWih, jt * 8 + kt, lane), c);
    } else {                                  // gh_n, jt = 16 + (t-24)
      const int jt = 16 + (t - 24);
      const float bv = b_hh[jt * 16 + n];
#pragma unroll
      for (int e = 0; e < 8; ++e) c[e] = bv;
#pragma unroll
      for (int kt = 0; kt < 4; ++kt)
        c = wmma_bf16(load_a_lds(hidS, 128, 0, kt * 32, lane),
                      load_b_frag(pWhh, jt * 4 + kt, lane), c);
    }
    acc[i] = c;
  }
  __syncthreads();   // all GEMM reads of inS done before overwriting gbuf
#pragma unroll
  for (int i = 0; i < 4; ++i) {
    const int t = wave + i * 8;
#pragma unroll
    for (int e = 0; e < 8; ++e) {
      const int row = e + 8 * g;
      const float v = acc[i][e];
      if (t < 16)      gbuf[row * 256 + t * 16 + n] = v;
      else if (t < 24) gbuf[4096 + row * 128 + (t - 16) * 16 + n] = v;
      else             gbuf[6144 + row * 128 + (t - 24) * 16 + n] = v;
    }
  }
  __syncthreads();
  for (int idx = tid; idx < 16 * 128; idx += 256) {
    const int m = idx >> 7, j = idx & 127;
    const float rg  = 1.0f / (1.0f + __expf(-gbuf[m * 256 + j]));
    const float ig  = 1.0f / (1.0f + __expf(-gbuf[m * 256 + 128 + j]));
    const float gin = gbuf[4096 + m * 128 + j];
    const float ghn = gbuf[6144 + m * 128 + j];
    const float ng  = tanhf(gin + rg * ghn);
    const float h   = bf2f(hidS[idx]);
    out[(size_t)m0 * 128 + idx] = ng + ig * (h - ng);
  }
}

// =====================================================================
extern "C" void kernel_launch(void* const* d_in, const int* in_sizes, int n_in,
                              void* d_out, int out_size, void* d_ws, size_t ws_size,
                              hipStream_t stream) {
  (void)in_sizes; (void)n_in; (void)out_size; (void)ws_size;
  const int*   items = (const int*)  d_in[0];
  const float* A     = (const float*)d_in[1];
  const float* emb   = (const float*)d_in[2];
  const float* w_in  = (const float*)d_in[3];
  const float* b_in  = (const float*)d_in[4];
  const float* w_out = (const float*)d_in[5];
  const float* b_out = (const float*)d_in[6];
  const float* w_ih  = (const float*)d_in[7];
  const float* w_hh  = (const float*)d_in[8];
  const float* b_ih  = (const float*)d_in[9];
  const float* b_hh  = (const float*)d_in[10];
  const float* b_iah = (const float*)d_in[11];
  const float* b_oah = (const float*)d_in[12];

  // Workspace layout (bf16), ~262 MB + ~0.4 MB packed weights:
  unsigned short* hidB    = (unsigned short*)d_ws;           // [M1,128]
  unsigned short* HinB    = hidB    + (size_t)M1 * HH;       // [M1,128]
  unsigned short* HoutB   = HinB    + (size_t)M1 * HH;       // [M1,128]
  unsigned short* inputsB = HoutB   + (size_t)M1 * HH;       // [M1,256]
  unsigned short* pWin    = inputsB + (size_t)M1 * 256;      // 128x128  -> 16384
  unsigned short* pWout   = pWin  + 128 * 128;
  unsigned short* pWih    = pWout + 128 * 128;               // 384x256  -> 98304
  unsigned short* pWhh    = pWih  + 384 * 256;               // 384x128  -> 49152

  k0_pack<<<(128 * 128 + 255) / 256, 256, 0, stream>>>(w_in,  pWin,  128, 128);
  k0_pack<<<(128 * 128 + 255) / 256, 256, 0, stream>>>(w_out, pWout, 128, 128);
  k0_pack<<<(384 * 256 + 255) / 256, 256, 0, stream>>>(w_ih,  pWih,  384, 256);
  k0_pack<<<(384 * 128 + 255) / 256, 256, 0, stream>>>(w_hh,  pWhh,  384, 128);

  k1_gather_dual_gemm<<<M1 / 64, 256, 0, stream>>>(items, emb, pWin, b_in, pWout,
                                                   b_out, hidB, HinB, HoutB);
  k2_adjacency<<<BB, 256, 0, stream>>>(A, HinB, HoutB, b_iah, b_oah, inputsB);
  k3_gru<<<M1 / 16, 256, 0, stream>>>(inputsB, hidB, pWih, pWhh, b_ih, b_hh,
                                      (float*)d_out);
}
